// TranslationModel_10007273799849
// MI455X (gfx1250) — matrix operand order repaired and checked by reading, not compiled, not run on hardware
//
#include <hip/hip_runtime.h>

// ---------------- constants (match reference) ----------------
constexpr int H      = 512;
constexpr int H3     = 3 * H;        // 1536
constexpr int MAXLEN = 500;
constexpr int L      = 400;
constexpr int T      = 400;
constexpr int TGT_V  = 32000;
constexpr int SOS    = 1;
constexpr int NB     = TGT_V / 256;  // 125 column blocks in output projection

// ---------------- WMMA types ----------------
typedef __attribute__((ext_vector_type(16))) __bf16 v16bf;
typedef __attribute__((ext_vector_type(8)))  float  v8f;

union BF16x16 { v16bf v; unsigned short u[16]; };

// ---------------- TDM descriptor types ----------------
typedef __attribute__((ext_vector_type(4))) unsigned tdm_v4u;
typedef __attribute__((ext_vector_type(8))) int      tdm_v8i;
typedef __attribute__((ext_vector_type(4))) int      tdm_v4i;

__device__ __forceinline__ unsigned short f32_to_bf16(float f) {
    union { float f; unsigned u; } x; x.f = f;
    unsigned r = (x.u + 0x7FFFu + ((x.u >> 16) & 1u)) >> 16;
    return (unsigned short)r;
}

__device__ __forceinline__ float wave_reduce_sum(float v) {
    // wave32 shuffle reduction
    #pragma unroll
    for (int off = 16; off > 0; off >>= 1) v += __shfl_down(v, off, 32);
    return v;
}

__device__ __forceinline__ float sigm(float x) { return 1.0f / (1.0f + expf(-x)); }

// Load a 16x32 bf16 A-fragment (row-major A[M][K]) in contiguous k order (k = 16*half + s), same as B.
__device__ __forceinline__ v16bf load_a_frag(const unsigned short* __restrict__ A,
                                             int mrow, int K, int k0, int half) {
    BF16x16 af;
    const unsigned short* Ar = A + (size_t)mrow * K + k0;
    #pragma unroll
    for (int v = 0; v < 8; ++v) {
        int kb = half * 16 + 2 * v;
        af.u[2 * v]     = Ar[kb];
        af.u[2 * v + 1] = Ar[kb + 1];
    }
    return af.v;
}

// Load a 32x16 bf16 B-fragment from weight W[N][K] row-major (logical B[k][n] = W[n][k]).
__device__ __forceinline__ v16bf load_b_frag(const unsigned short* __restrict__ W,
                                             int ncol, int K, int k0, int half) {
    BF16x16 bf;
    const unsigned short* Wr = W + (size_t)ncol * K + k0;
    #pragma unroll
    for (int v = 0; v < 8; ++v) {
        int kb = half * 16 + 2 * v;
        bf.u[2 * v]     = Wr[kb];
        bf.u[2 * v + 1] = Wr[kb + 1];
    }
    return bf.v;
}

// ---------------- Tensor Data Mover: 2-D bf16 tile (tile_d1 rows x tile_d0 elems) -> LDS ----------------
// D# built per CDNA5 ISA.
__device__ __forceinline__ void tdm_load_2d_bf16(unsigned lds_off,
                                                 const unsigned short* __restrict__ gsrc,
                                                 unsigned tensor_d0, unsigned tensor_d1,
                                                 unsigned tile_d0, unsigned tile_d1,
                                                 unsigned long long stride0) {
    unsigned long long ga = (unsigned long long)(const void*)gsrc;
    tdm_v4u g0;
    g0.x = 1u;                                                   // count=1 (valid), user mode
    g0.y = lds_off;                                              // LDS byte address
    g0.z = (unsigned)(ga & 0xFFFFFFFFu);                         // global_addr[31:0]
    g0.w = (unsigned)((ga >> 32) & 0x01FFFFFFu) | (2u << 30);    // global_addr[56:32] | type=2
    tdm_v8i g1;
    g1[0] = (int)(1u << 16);                                     // data_size=1 (2B), no multicast
    g1[1] = (int)((tensor_d0 & 0xFFFFu) << 16);                  // tensor_dim0[15:0] @ bits 63:48
    g1[2] = (int)((tensor_d0 >> 16) | ((tensor_d1 & 0xFFFFu) << 16));
    g1[3] = (int)((tensor_d1 >> 16) | (tile_d0 << 16));          // tile_dim0 @ bits 127:112
    g1[4] = (int)(tile_d1 & 0xFFFFu);                            // tile_dim1 (tile_dim2 = 0)
    g1[5] = (int)(stride0 & 0xFFFFFFFFull);                      // tensor_dim0_stride[31:0]
    g1[6] = (int)((stride0 >> 32) & 0xFFFFull);                  // stride0[47:32]; stride1 = 0
    g1[7] = 0;
    tdm_v4i gz = {0, 0, 0, 0};
#if defined(__clang_major__) && (__clang_major__ >= 23)
    tdm_v8i gz8 = {0, 0, 0, 0, 0, 0, 0, 0};
    __builtin_amdgcn_tensor_load_to_lds(g0, g1, gz, gz, gz8, 0);
#else
    __builtin_amdgcn_tensor_load_to_lds(g0, g1, gz, gz, 0);
#endif
}

// ---------------- kernel 1: f32 -> bf16 conversion ----------------
__global__ void k_cvt_bf16(const float* __restrict__ s, unsigned short* __restrict__ d, long n) {
    long i = (long)blockIdx.x * blockDim.x + threadIdx.x;
    if (i < n) d[i] = f32_to_bf16(s[i]);
}

// ---------------- kernel 2: gather token embeddings -> bf16 rows ----------------
__global__ void k_gather_emb(const int* __restrict__ toks, const float* __restrict__ emb,
                             unsigned short* __restrict__ A) {
    int t = blockIdx.x;
    int tok = toks[t];
    for (int i = threadIdx.x; i < H; i += blockDim.x)
        A[(size_t)t * H + i] = f32_to_bf16(emb[(size_t)tok * H + i]);
}

// ---------------- kernel 3: generic WMMA GEMM  C[M,N] = A[M,K] * W[N,K]^T + bias ----------------
// grid = (M/16, N/64), block = 128 (4 waves, one 16x16 N-tile each)
__global__ void k_gemm_bias(const unsigned short* __restrict__ A,
                            const unsigned short* __restrict__ W,
                            const float* __restrict__ bias,
                            float* __restrict__ C, int Nr, int Kr) {
    int tid = threadIdx.x, lane = tid & 31, wv = tid >> 5, half = lane >> 4;
    int m0 = blockIdx.x * 16;
    int n0 = blockIdx.y * 64 + wv * 16;
    int mrow = m0 + (lane & 15);
    int ncol = n0 + (lane & 15);

    v8f acc;
    #pragma unroll
    for (int r = 0; r < 8; ++r) acc[r] = 0.0f;

    for (int kc = 0; kc < Kr; kc += 32) {
        v16bf a = load_a_frag(A, mrow, Kr, kc, half);
        v16bf b = load_b_frag(W, ncol, Kr, kc, half);
        acc = __builtin_amdgcn_wmma_f32_16x16x32_bf16(
                  false, a, false, b, (short)0, acc, false, false);
    }
    #pragma unroll
    for (int r = 0; r < 8; ++r) {
        int row = m0 + r + 8 * half;
        C[(size_t)row * Nr + ncol] = acc[r] + bias[ncol];
    }
}

// ---------------- wave-cooperative matvec:  out[N] = W[N,K] * vec[K] + bias ----------------
__device__ void matvec_wave(const float* __restrict__ W, const float* __restrict__ vec,
                            const float* __restrict__ bias, float* __restrict__ out,
                            int N, int K, int tid, int nthreads) {
    int lane = tid & 31, wv = tid >> 5, nw = nthreads >> 5;
    for (int o = wv; o < N; o += nw) {
        const float* Wr = W + (size_t)o * K;
        float p = 0.0f;
        for (int k = lane; k < K; k += 32) p += Wr[k] * vec[k];
        p = wave_reduce_sum(p);
        if (lane == 0) out[o] = p + bias[o];
    }
}

// ---------------- kernel 4: encoder recurrence (single persistent WGP) ----------------
__global__ void k_enc(const float* __restrict__ GI, const float* __restrict__ Whh,
                      const float* __restrict__ bhh, float* __restrict__ enc_out) {
    __shared__ float h[H];
    __shared__ float gh[H3];
    int tid = threadIdx.x;
    for (int i = tid; i < H; i += blockDim.x) h[i] = 0.0f;
    // zero-pad rows [L, MAXLEN)
    for (int i = L * H + tid; i < MAXLEN * H; i += blockDim.x) enc_out[i] = 0.0f;
    __syncthreads();

    for (int t = 0; t < L; ++t) {
        matvec_wave(Whh, h, bhh, gh, H3, H, tid, blockDim.x);
        __syncthreads();
        const float* gi = GI + (size_t)t * H3;
        if (tid < H) {
            float r  = sigm(gi[tid] + gh[tid]);
            float z  = sigm(gi[H + tid] + gh[H + tid]);
            float n  = tanhf(gi[2 * H + tid] + r * gh[2 * H + tid]);
            float hn = (1.0f - z) * n + z * h[tid];
            enc_out[(size_t)t * H + tid] = hn;
            h[tid] = hn;
        }
        __syncthreads();
    }
}

// ---------------- kernel 5: decoder recurrence with attention (single persistent WGP) ----------------
__global__ void k_dec(const int* __restrict__ y, const float* __restrict__ emb_tgt,
                      const float* __restrict__ attn_W, const float* __restrict__ attn_b,
                      const float* __restrict__ comb_W, const float* __restrict__ comb_b,
                      const float* __restrict__ Wih, const float* __restrict__ Whh,
                      const float* __restrict__ bih, const float* __restrict__ bhh,
                      const float* __restrict__ enc_out, unsigned short* __restrict__ H2b) {
    __shared__ float h[H];
    __shared__ float cat[2 * H];     // [e, h] then [e, ctx]
    __shared__ float a[MAXLEN];
    __shared__ float o[H];
    __shared__ float g[2 * H3];      // gi | gh
    __shared__ float red[1024];
    int tid = threadIdx.x;
    for (int i = tid; i < H; i += blockDim.x) h[i] = 0.0f;
    __syncthreads();

    for (int t = 0; t < T; ++t) {
        int tok = (t == 0) ? SOS : y[t - 1];
        for (int i = tid; i < H; i += blockDim.x) {
            cat[i]     = emb_tgt[(size_t)tok * H + i];  // e
            cat[H + i] = h[i];
        }
        __syncthreads();

        // attention logits over MAXLEN
        matvec_wave(attn_W, cat, attn_b, a, MAXLEN, 2 * H, tid, blockDim.x);
        __syncthreads();

        // softmax over a[0..MAXLEN)
        red[tid] = (tid < MAXLEN) ? a[tid] : -1e30f;
        __syncthreads();
        for (int s = 512; s >= 1; s >>= 1) {
            if (tid < s) red[tid] = fmaxf(red[tid], red[tid + s]);
            __syncthreads();
        }
        float mx = red[0];
        __syncthreads();
        if (tid < MAXLEN) a[tid] = expf(a[tid] - mx);
        red[tid] = (tid < MAXLEN) ? a[tid] : 0.0f;
        __syncthreads();
        for (int s = 512; s >= 1; s >>= 1) {
            if (tid < s) red[tid] += red[tid + s];
            __syncthreads();
        }
        float sm = red[0];
        __syncthreads();
        if (tid < MAXLEN) a[tid] /= sm;
        __syncthreads();

        // ctx = a @ enc_out  -> cat[H..2H)
        for (int j = tid; j < H; j += blockDim.x) {
            float c = 0.0f;
            for (int i = 0; i < MAXLEN; ++i) c += a[i] * enc_out[(size_t)i * H + j];
            cat[H + j] = c;
        }
        __syncthreads();

        // o = relu(comb_W @ [e, ctx] + comb_b)
        matvec_wave(comb_W, cat, comb_b, o, H, 2 * H, tid, blockDim.x);
        __syncthreads();
        if (tid < H) o[tid] = fmaxf(o[tid], 0.0f);
        __syncthreads();

        // GRU gates
        matvec_wave(Wih, o, bih, g, H3, H, tid, blockDim.x);        // gi
        matvec_wave(Whh, h, bhh, g + H3, H3, H, tid, blockDim.x);   // gh
        __syncthreads();
        if (tid < H) {
            float r  = sigm(g[tid] + g[H3 + tid]);
            float z  = sigm(g[H + tid] + g[H3 + H + tid]);
            float n  = tanhf(g[2 * H + tid] + r * g[H3 + 2 * H + tid]);
            float hn = (1.0f - z) * n + z * h[tid];
            H2b[(size_t)t * H + tid] = f32_to_bf16(hn);
            h[tid] = hn;
        }
        __syncthreads();
    }
}

// ---------------- kernel 6: fused output projection + blockwise softmax stats ----------------
// grid = (T/16, NB), block = 256 (8 waves x two 16x16 N-tiles = 256 columns).
// B panel is staged through LDS by the Tensor Data Mover (double buffered, one
// tensor_load_to_lds per 256x32 bf16 K-slab), overlapped with WMMA on the
// previous slab and synchronized with s_wait_tensorcnt + workgroup barrier.
__global__ void k_outproj(const unsigned short* __restrict__ H2b,
                          const unsigned short* __restrict__ Wb,
                          const float* __restrict__ ob, const int* __restrict__ y,
                          float* __restrict__ mblk, float* __restrict__ sblk,
                          float* __restrict__ logity) {
    __shared__ float tile[16 * 256];
    __shared__ unsigned short bufB[2][256 * 32];   // 2 x 16KB bf16 K-slabs
    int tid = threadIdx.x, lane = tid & 31, wv = tid >> 5, half = lane >> 4;
    int t0 = blockIdx.x * 16;
    int c0 = blockIdx.y * 256;
    int mrow = t0 + (lane & 15);
    const int KSTEPS = H / 32;     // 16

    // issue TDM for K-slab 0 (whole 256-row x 32-elem bf16 tile of out_W panel)
    if (wv == 0)
        tdm_load_2d_bf16((unsigned)(size_t)&bufB[0][0],
                         Wb + (size_t)c0 * H,
                         /*tensor_d0=*/H, /*tensor_d1=*/TGT_V,
                         /*tile_d0=*/32, /*tile_d1=*/256,
                         /*stride0=*/H);

    v8f acc0, acc1;
    #pragma unroll
    for (int r = 0; r < 8; ++r) { acc0[r] = 0.0f; acc1[r] = 0.0f; }

    for (int kc = 0; kc < KSTEPS; ++kc) {
        if (wv == 0) __builtin_amdgcn_s_wait_tensorcnt(0);
        __syncthreads();                       // slab kc visible to all waves
        if (wv == 0 && kc + 1 < KSTEPS)        // overlap next DMA with compute
            tdm_load_2d_bf16((unsigned)(size_t)&bufB[(kc + 1) & 1][0],
                             Wb + (size_t)c0 * H + (kc + 1) * 32,
                             H, TGT_V, 32, 256, H);

        // prefetch next A slab (tiny; stays hot in L0 across all 8 waves)
        if (kc + 1 < KSTEPS)
            __builtin_prefetch(H2b + (size_t)mrow * H + (kc + 1) * 32, 0, 3);

        v16bf a = load_a_frag(H2b, mrow, H, kc * 32, half);

        const unsigned short* Bl = &bufB[kc & 1][0];
        int nl0 = wv * 32 + (lane & 15);       // local row (column of C) for tile 0
        BF16x16 b0, b1;
        #pragma unroll
        for (int v = 0; v < 8; ++v) {
            int kb = half * 16 + 2 * v;
            b0.u[2 * v]     = Bl[nl0 * 32 + kb];
            b0.u[2 * v + 1] = Bl[nl0 * 32 + kb + 1];
            b1.u[2 * v]     = Bl[(nl0 + 16) * 32 + kb];
            b1.u[2 * v + 1] = Bl[(nl0 + 16) * 32 + kb + 1];
        }
        acc0 = __builtin_amdgcn_wmma_f32_16x16x32_bf16(false, a, false, b0.v,
                                                       (short)0, acc0, false, false);
        acc1 = __builtin_amdgcn_wmma_f32_16x16x32_bf16(false, a, false, b1.v,
                                                       (short)0, acc1, false, false);
    }

    int col = wv * 32 + (lane & 15);
    #pragma unroll
    for (int r = 0; r < 8; ++r) {
        int row = r + 8 * half;
        tile[row * 256 + col]      = acc0[r] + ob[c0 + col];
        tile[row * 256 + col + 16] = acc1[r] + ob[c0 + col + 16];
    }
    __syncthreads();

    if (tid < 16) {
        int t = t0 + tid;
        const float* tr = tile + tid * 256;
        float M = -1e30f;
        for (int c = 0; c < 256; ++c) M = fmaxf(M, tr[c]);
        float S = 0.0f;
        for (int c = 0; c < 256; ++c) S += expf(tr[c] - M);
        mblk[(size_t)t * NB + blockIdx.y] = M;
        sblk[(size_t)t * NB + blockIdx.y] = S;
        int yt = y[t];
        if (yt >= c0 && yt < c0 + 256) logity[t] = tr[yt - c0];
    }
}

// ---------------- kernel 7: combine per-block stats into mean NLL ----------------
__global__ void k_final(const float* __restrict__ mblk, const float* __restrict__ sblk,
                        const float* __restrict__ logity, float* __restrict__ out) {
    __shared__ float red[512];
    int tid = threadIdx.x;
    float acc = 0.0f;
    for (int t = tid; t < T; t += blockDim.x) {
        float M = -1e30f;
        for (int b = 0; b < NB; ++b) M = fmaxf(M, mblk[(size_t)t * NB + b]);
        float S = 0.0f;
        for (int b = 0; b < NB; ++b) S += sblk[(size_t)t * NB + b] * expf(mblk[(size_t)t * NB + b] - M);
        acc += -(logity[t] - M - logf(S));
    }
    red[tid] = acc;
    __syncthreads();
    for (int s = 256; s >= 1; s >>= 1) {
        if (tid < s) red[tid] += red[tid + s];
        __syncthreads();
    }
    if (tid == 0) out[0] = red[0] / (float)T;
}

// ---------------- launch ----------------
extern "C" void kernel_launch(void* const* d_in, const int* in_sizes, int n_in,
                              void* d_out, int out_size, void* d_ws, size_t ws_size,
                              hipStream_t stream) {
    const int*   x        = (const int*)  d_in[0];
    const int*   y        = (const int*)  d_in[1];
    const float* emb_src  = (const float*)d_in[2];
    const float* enc_Wih  = (const float*)d_in[3];
    const float* enc_Whh  = (const float*)d_in[4];
    const float* enc_bih  = (const float*)d_in[5];
    const float* enc_bhh  = (const float*)d_in[6];
    const float* emb_tgt  = (const float*)d_in[7];
    const float* attn_W   = (const float*)d_in[8];
    const float* attn_b   = (const float*)d_in[9];
    const float* comb_W   = (const float*)d_in[10];
    const float* comb_b   = (const float*)d_in[11];
    const float* dec_Wih  = (const float*)d_in[12];
    const float* dec_Whh  = (const float*)d_in[13];
    const float* dec_bih  = (const float*)d_in[14];
    const float* dec_bhh  = (const float*)d_in[15];
    const float* out_W    = (const float*)d_in[16];
    const float* out_b    = (const float*)d_in[17];

    // workspace carve-out (256B aligned slices)
    char* base = (char*)d_ws;
    size_t off = 0;
    auto take = [&](size_t bytes) -> void* {
        off = (off + 255) & ~(size_t)255;
        void* p = base + off;
        off += bytes;
        return p;
    };
    float*          ws_enc_out = (float*)         take((size_t)MAXLEN * H * 4);
    float*          ws_GI      = (float*)         take((size_t)L * H3 * 4);
    unsigned short* ws_Aemb    = (unsigned short*)take((size_t)L * H * 2);
    unsigned short* ws_Wih_b   = (unsigned short*)take((size_t)H3 * H * 2);
    unsigned short* ws_outW_b  = (unsigned short*)take((size_t)TGT_V * H * 2);
    unsigned short* ws_H2b     = (unsigned short*)take((size_t)T * H * 2);
    float*          ws_mblk    = (float*)         take((size_t)T * NB * 4);
    float*          ws_sblk    = (float*)         take((size_t)T * NB * 4);
    float*          ws_ly      = (float*)         take((size_t)T * 4);
    (void)ws_size; (void)in_sizes; (void)n_in; (void)out_size;

    const long n_wih = (long)H3 * H;          // 786,432
    const long n_ow  = (long)TGT_V * H;       // 16,384,000

    // 1-2) weight conversion to bf16
    k_cvt_bf16<<<dim3((unsigned)((n_wih + 255) / 256)), 256, 0, stream>>>(enc_Wih, ws_Wih_b, n_wih);
    k_cvt_bf16<<<dim3((unsigned)((n_ow + 255) / 256)),  256, 0, stream>>>(out_W,  ws_outW_b, n_ow);

    // 3) gather source embeddings (bf16 rows)
    k_gather_emb<<<dim3(L), 256, 0, stream>>>(x, emb_src, ws_Aemb);

    // 4) encoder input gates as one WMMA GEMM: GI[400,1536] = Aemb * enc_Wih^T + bih
    k_gemm_bias<<<dim3(L / 16, H3 / 64), 128, 0, stream>>>(ws_Aemb, ws_Wih_b, enc_bih, ws_GI, H3, H);

    // 5) encoder recurrence (one persistent workgroup)
    k_enc<<<dim3(1), 1024, 0, stream>>>(ws_GI, enc_Whh, enc_bhh, ws_enc_out);

    // 6) decoder recurrence with attention (one persistent workgroup)
    k_dec<<<dim3(1), 1024, 0, stream>>>(y, emb_tgt, attn_W, attn_b, comb_W, comb_b,
                                        dec_Wih, dec_Whh, dec_bih, dec_bhh,
                                        ws_enc_out, ws_H2b);

    // 7) fused output projection (WMMA + TDM-staged B panel) + blockwise softmax stats
    k_outproj<<<dim3(T / 16, NB), 256, 0, stream>>>(ws_H2b, ws_outW_b, out_b, y,
                                                    ws_mblk, ws_sblk, ws_ly);

    // 8) combine into mean NLL
    k_final<<<dim3(1), 512, 0, stream>>>(ws_mblk, ws_sblk, ws_ly, (float*)d_out);
}
